// GridNetBlock_57037165690891
// MI455X (gfx1250) — compile-verified, matrix operand order checked by reading, not run on hardware
//
#include <hip/hip_runtime.h>
#include <math.h>

// ---------------- problem constants ----------------
#define B_    4
#define T_    512
#define NF_   65
#define C_    128
#define NH_   4
#define E_    8
#define H_    256
#define DOWN_ 4
#define L_    100
#define VD_   32
#define QD_   16      // NF/DOWN
#define BT_   2048    // B*T
#define BNF_  260     // B*NF
#define S_    611     // T + L - 1
#define TOT_  ((long)B_*T_*NF_*C_)   // 17,039,360

typedef __bf16 bf16_t;
typedef bf16_t v16bf __attribute__((ext_vector_type(16)));
typedef bf16_t v8bf  __attribute__((ext_vector_type(8)));
typedef float  v8f   __attribute__((ext_vector_type(8)));
typedef float  v4f   __attribute__((ext_vector_type(4)));

__device__ __forceinline__ float sigf(float x) { return 1.f / (1.f + __expf(-x)); }

// two-value block reduction (256 threads); leaves sums in s1/s2 for all threads
#define REDUCE2_256(s1, s2)                                                    \
  __shared__ float _r1[256], _r2[256];                                         \
  _r1[threadIdx.x] = (s1); _r2[threadIdx.x] = (s2); __syncthreads();           \
  for (int _o = 128; _o > 0; _o >>= 1) {                                       \
    if ((int)threadIdx.x < _o) { _r1[threadIdx.x] += _r1[threadIdx.x + _o];    \
                                 _r2[threadIdx.x] += _r2[threadIdx.x + _o]; }  \
    __syncthreads();                                                           \
  }                                                                            \
  (s1) = _r1[0]; (s2) = _r2[0];

// ---------------- WMMA fragment loaders (branch-free) ----------------
// Fragment element j <-> k = k0 + hs*8 + j + (j>=8 ? 8 : 0)   (ISA 16-bit A 16x32 layout)

// f32 source, whole 32-chunk in K bounds, ld % 4 == 0; row pointer pre-clamped
__device__ __forceinline__ v16bf ldfrag_vec(const float* __restrict__ p) {
  v4f a = *(const v4f*)(p);
  v4f b = *(const v4f*)(p + 4);
  v4f c = *(const v4f*)(p + 16);
  v4f d = *(const v4f*)(p + 20);
  v16bf fr;
#pragma unroll
  for (int j = 0; j < 4; ++j) {
    fr[j]      = (bf16_t)a[j];
    fr[4 + j]  = (bf16_t)b[j];
    fr[8 + j]  = (bf16_t)c[j];
    fr[12 + j] = (bf16_t)d[j];
  }
  return fr;
}
// f32 scalar path with K guard (address clamped, value zeroed past K); row pre-clamped
__device__ __forceinline__ v16bf ldfrag_row(const float* __restrict__ p, int k0, int koff, int K) {
  v16bf fr;
#pragma unroll
  for (int j = 0; j < 16; ++j) {
    int k = k0 + koff + j + ((j & 8) ? 8 : 0);
    float v = p[k < K ? k : 0];
    fr[j] = (bf16_t)(k < K ? v : 0.f);
  }
  return fr;
}
// f32 B stored [K,N]: fixed column per lane (pcol = B + clamped_col), stride ld over k
__device__ __forceinline__ v16bf ldfrag_col(const float* __restrict__ pcol, long ld,
                                            int k0, int koff, int K) {
  v16bf fr;
#pragma unroll
  for (int j = 0; j < 16; ++j) {
    int k = k0 + koff + j + ((j & 8) ? 8 : 0);
    float v = pcol[(long)(k < K ? k : 0) * ld];
    fr[j] = (bf16_t)(k < K ? v : 0.f);
  }
  return fr;
}
// pre-packed bf16 weights [N,K]: two 16-byte loads, zero converts
__device__ __forceinline__ v16bf ldfrag_h(const bf16_t* __restrict__ p) {
  v8bf lo = *(const v8bf*)(p);
  v8bf hi = *(const v8bf*)(p + 16);
  v16bf fr;
#pragma unroll
  for (int j = 0; j < 8; ++j) { fr[j] = lo[j]; fr[8 + j] = hi[j]; }
  return fr;
}
__device__ __forceinline__ v16bf ldfrag_h_g(const bf16_t* __restrict__ p, int k0, int koff, int K) {
  v16bf fr;
#pragma unroll
  for (int j = 0; j < 16; ++j) {
    int k = k0 + koff + j + ((j & 8) ? 8 : 0);
    bf16_t v = p[k < K ? k : 0];
    fr[j] = (k < K) ? v : (bf16_t)0.f;
  }
  return fr;
}

// ---------------------------------------------------------------------------
// Generic bf16 WMMA GEMM:  C[M,N] = A[M,K] @ B  (+bias, +PReLU, +=C)
// MODE: 0 = f32 B^T (scores);         1 = f32 B [K,N], scalar A (probs@Vf);
//       2 = bf16 B^T plain;           3 = bf16 B^T + bias;
//       4 = bf16 B^T + bias + PReLU;  5 = bf16 B^T accumulate (C += A@B^T)
// One wave computes 64x16 (4 sub-tiles along M share one B fragment).
// Out-of-range rows/cols are address-clamped only (their outputs are never
// stored); only K-overrun is zeroed, in a separate scalar tail chunk.
// ---------------------------------------------------------------------------
template<int MODE>
__global__ __launch_bounds__(128)
void gemm_wmma(const float* __restrict__ A, long lda, long sA,
               const void* __restrict__ Bmv, long ldb, long sB,
               float* __restrict__ Cm, long ldc, long sC,
               int M, int N, int K,
               const float* __restrict__ bias,
               const float* __restrict__ pa)
{
  constexpr bool BT_L   = (MODE != 1);
  constexpr bool ACC_   = (MODE == 5);
  constexpr bool BIAS_  = (MODE == 3 || MODE == 4);
  constexpr bool PRELU_ = (MODE == 4);
  constexpr bool VECL   = (MODE != 1);   // 128-bit A loads (lda%4==0)
  constexpr bool B16    = (MODE >= 2);   // pre-packed bf16 B

  const int tiles_n = (N + 15) >> 4;
  const int tiles_m = (M + 63) >> 6;
  const int wid = blockIdx.x * 4 + ((int)threadIdx.x >> 5);
  if (wid >= tiles_m * tiles_n) return;
  const int tm = wid / tiles_n, tn = wid - tm * tiles_n;
  const int mb = tm << 6, nb = tn << 4;
  const int lane = (int)threadIdx.x & 31;
  const int rl = lane & 15;
  const int hs = lane >> 4;
  const int koff = hs * 8;

  A  += (long)blockIdx.z * sA;
  Cm += (long)blockIdx.z * sC;
  const float*  Bf = (const float*)Bmv  + (long)blockIdx.z * sB;
  const bf16_t* Bh = (const bf16_t*)Bmv + (long)blockIdx.z * sB;

  // clamped per-lane row/col base pointers
  const float* Arow[4];
#pragma unroll
  for (int u = 0; u < 4; ++u) {
    int m = mb + u * 16 + rl;
    Arow[u] = A + (long)(m < M ? m : 0) * lda;
  }
  const int ncl = (nb + rl) < N ? (nb + rl) : 0;
  const float*  Browf = BT_L ? (Bf + (long)ncl * ldb) : (Bf + ncl);
  const bf16_t* Browh = Bh + (long)ncl * ldb;

  v8f acc[4];
#pragma unroll
  for (int u = 0; u < 4; ++u) {
#pragma unroll
    for (int r = 0; r < 8; ++r) {
      if (ACC_) {
        int m = mb + u * 16 + r + hs * 8, n = nb + rl;
        acc[u][r] = (m < M && n < N) ? Cm[(long)m * ldc + n] : 0.f;
      } else {
        acc[u][r] = 0.f;
      }
    }
  }

  const int Kfull = K & ~31;
  for (int k0 = 0; k0 < Kfull; k0 += 32) {
    v16bf bfr;
    if (B16)        bfr = ldfrag_h(Browh + k0 + koff);
    else if (BT_L)  bfr = ldfrag_vec(Browf + k0 + koff);
    else            bfr = ldfrag_col(Browf, ldb, k0, koff, K);
#pragma unroll
    for (int u = 0; u < 4; ++u) {
      v16bf afr;
      if (VECL) afr = ldfrag_vec(Arow[u] + k0 + koff);
      else      afr = ldfrag_row(Arow[u], k0, koff, K);
      acc[u] = __builtin_amdgcn_wmma_f32_16x16x32_bf16(
          false, afr, false, bfr, (short)0, acc[u], false, false);
    }
  }
  if (Kfull < K) {           // ragged-K tail (scalar, zero-padded past K)
    v16bf bfr;
    if (B16)        bfr = ldfrag_h_g(Browh, Kfull, koff, K);
    else if (BT_L)  bfr = ldfrag_row(Browf, Kfull, koff, K);
    else            bfr = ldfrag_col(Browf, ldb, Kfull, koff, K);
#pragma unroll
    for (int u = 0; u < 4; ++u) {
      v16bf afr = ldfrag_row(Arow[u], Kfull, koff, K);
      acc[u] = __builtin_amdgcn_wmma_f32_16x16x32_bf16(
          false, afr, false, bfr, (short)0, acc[u], false, false);
    }
  }

  float a0 = PRELU_ ? pa[0] : 0.f;
#pragma unroll
  for (int u = 0; u < 4; ++u) {
#pragma unroll
    for (int r = 0; r < 8; ++r) {
      int m = mb + u * 16 + r + hs * 8, n = nb + rl;
      if (m < M && n < N) {
        float v = acc[u][r];
        if (BIAS_)  v += bias[n];
        if (PRELU_) v = v >= 0.f ? v : a0 * v;
        Cm[(long)m * ldc + n] = v;
      }
    }
  }
}

static inline void gemm(const float* A, long lda, long sA,
                        const void* Bm, long ldb, long sB,
                        float* Cm, long ldc, long sC,
                        int M, int N, int K, int batch,
                        const float* bias, const float* pa,
                        int mode, hipStream_t st)
{
  int tiles = ((M + 63) >> 6) * ((N + 15) >> 4);
  dim3 g((unsigned)((tiles + 3) >> 2), 1, (unsigned)batch), b(128);
  switch (mode) {
    case 0: gemm_wmma<0><<<g, b, 0, st>>>(A, lda, sA, Bm, ldb, sB, Cm, ldc, sC, M, N, K, bias, pa); break;
    case 1: gemm_wmma<1><<<g, b, 0, st>>>(A, lda, sA, Bm, ldb, sB, Cm, ldc, sC, M, N, K, bias, pa); break;
    case 2: gemm_wmma<2><<<g, b, 0, st>>>(A, lda, sA, Bm, ldb, sB, Cm, ldc, sC, M, N, K, bias, pa); break;
    case 3: gemm_wmma<3><<<g, b, 0, st>>>(A, lda, sA, Bm, ldb, sB, Cm, ldc, sC, M, N, K, bias, pa); break;
    case 4: gemm_wmma<4><<<g, b, 0, st>>>(A, lda, sA, Bm, ldb, sB, Cm, ldc, sC, M, N, K, bias, pa); break;
    default: gemm_wmma<5><<<g, b, 0, st>>>(A, lda, sA, Bm, ldb, sB, Cm, ldc, sC, M, N, K, bias, pa); break;
  }
}

// ---------------- small utility kernels ----------------
__global__ void k_fill0(float* p, long n) {
  long i = (long)blockIdx.x * 256 + threadIdx.x;
  if (i < n) p[i] = 0.f;
}
__global__ void k_copy(float* d, const float* s, long n) {
  long i = (long)blockIdx.x * 256 + threadIdx.x;
  if (i < n) d[i] = s[i];
}
__global__ void k_cvt_bf16(bf16_t* __restrict__ d, const float* __restrict__ s, long n) {
  long i = (long)blockIdx.x * 256 + threadIdx.x;
  if (i < n) d[i] = (bf16_t)s[i];
}

// pack x -> A_conv [32768, DOWN*C]: A[(n,i)][m*C+c] = x[n, i*4+m, c]
__global__ void k_pack_aconv(const float* __restrict__ x, float* __restrict__ A) {
  long i = (long)blockIdx.x * 256 + threadIdx.x;
  if (i >= (long)32768 * 512) return;
  int col = (int)(i & 511); long r = i >> 9;
  int ii = (int)(r & 15); long n = r >> 4;
  int m = col >> 7, c = col & 127;
  A[i] = x[n * (long)(NF_ * C_) + (long)(ii * DOWN_ + m) * C_ + c];
}
// conv_w[C,C,DOWN] -> bf16 Wc[o][m*C+c]
__global__ void k_pack_wc(const float* __restrict__ w, bf16_t* __restrict__ Wc) {
  int i = blockIdx.x * 256 + threadIdx.x;
  if (i >= 128 * 512) return;
  int col = i & 511, oo = i >> 9;
  int m = col >> 7, c = col & 127;
  Wc[i] = (bf16_t)w[oo * 512 + c * 4 + m];
}
// deconv_w[2H,C,DOWN] -> bf16 Wd[2][m*128+o][d(256)]
__global__ void k_pack_wd(const float* __restrict__ w, bf16_t* __restrict__ Wd) {
  int i = blockIdx.x * 256 + threadIdx.x;
  if (i >= 2 * 512 * 256) return;
  int half = i >> 17;          // 512*256 = 2^17
  int rem = i & 131071;
  int row = rem >> 8, dd = rem & 255;
  int m = row >> 7, oo = row & 127;
  int d = half * 256 + dd;
  Wd[i] = (bf16_t)w[d * 512 + oo * 4 + m];
}
__global__ void k_pack_whh(const float* __restrict__ wf, const float* __restrict__ wr,
                           bf16_t* __restrict__ out) {
  int i = blockIdx.x * 256 + threadIdx.x;
  if (i >= 2 * 1024 * 256) return;
  int dir = i >> 18; int rem = i & 262143;  // 1024*256 = 2^18
  out[i] = (bf16_t)(dir ? wr[rem] : wf[rem]);
}
__global__ void k_bias_sum(const float* a0, const float* b0, const float* a1,
                           const float* b1, const float* a2, const float* b2,
                           float* out) {
  int i = blockIdx.x * 256 + threadIdx.x;
  if (i >= 3072) return;
  int w = i >> 10, j = i & 1023;
  const float* A = (w == 0) ? a0 : (w == 1) ? a1 : a2;
  const float* Bv = (w == 0) ? b0 : (w == 1) ? b1 : b2;
  out[i] = A[j] + Bv[j];
}

// ---------------- LayerNorm kernels ----------------
// in-place LN over D per row; grid.x = rows, 256 threads
__global__ void k_ln_rows(float* __restrict__ x, const float* __restrict__ g,
                          const float* __restrict__ bb, int D) {
  long r = blockIdx.x;
  float* row = x + r * (long)D;
  float s1 = 0.f, s2 = 0.f;
  for (int j = threadIdx.x; j < D; j += 256) { float v = row[j]; s1 += v; s2 += v * v; }
  REDUCE2_256(s1, s2);
  float mean = s1 / D, var = s2 / D - mean * mean;
  float rstd = rsqrtf(var + 1e-5f);
  for (int j = threadIdx.x; j < D; j += 256) {
    float v = row[j];
    row[j] = (v - mean) * rstd * g[j] + bb[j];
  }
}
// LN over C of intra[b,t,f,:] then write transposed into yln[(b*NF+f), t, :]
__global__ void k_ln1t(const float* __restrict__ intra, float* __restrict__ yln,
                       const float* __restrict__ g, const float* __restrict__ bb) {
  int r = blockIdx.x;                  // (b*512+t)*65 + f
  int f = r % NF_; int bt = r / NF_;
  int t = bt & 511; int b = bt >> 9;
  const float* row = intra + (long)r * C_;
  float* drow = yln + (((long)(b * NF_ + f) * T_) + t) * C_;
  float s1 = 0.f, s2 = 0.f;
  for (int j = threadIdx.x; j < C_; j += 256) { float v = row[j]; s1 += v; s2 += v * v; }
  REDUCE2_256(s1, s2);
  float mean = s1 / C_, var = s2 / C_ - mean * mean;
  float rstd = rsqrtf(var + 1e-5f);
  for (int j = threadIdx.x; j < C_; j += 256) {
    float v = row[j];
    drow[j] = (v - mean) * rstd * g[j] + bb[j];
  }
}
// gather head (b,nh,t), LN over NF*d, write dst[bh, toff+t, f*d+e]
__global__ void k_ln_head(const float* __restrict__ src, float* __restrict__ dst,
                          const float* __restrict__ g, const float* __restrict__ bb,
                          int d, int toff, int Tdst) {
  int blk = blockIdx.x;                // bh*512 + t
  int bh = blk >> 9, t = blk & 511;
  int b = bh >> 2, nh = bh & 3;
  int D = NF_ * d;
  float s1 = 0.f, s2 = 0.f;
  for (int j = threadIdx.x; j < D; j += 256) {
    int f = j / d, e = j - f * d;
    float v = src[(((long)(b * T_ + t)) * NF_ + f) * (long)(NH_ * d) + nh * d + e];
    s1 += v; s2 += v * v;
  }
  REDUCE2_256(s1, s2);
  float mean = s1 / D, var = s2 / D - mean * mean;
  float rstd = rsqrtf(var + 1e-5f);
  long base = ((long)bh * Tdst + (toff + t)) * (long)D;
  for (int j = threadIdx.x; j < D; j += 256) {
    int f = j / d, e = j - f * d;
    float v = src[(((long)(b * T_ + t)) * NF_ + f) * (long)(NH_ * d) + nh * d + e];
    dst[base + j] = (v - mean) * rstd * g[j] + bb[j];
  }
}
// final: out[bt,:] = LN_{8320}(ph[bt,:]) * g + b + out[bt,:]   (out == inter)
__global__ void k_ln_final(const float* __restrict__ ph, float* __restrict__ out,
                           const float* __restrict__ g, const float* __restrict__ bb) {
  int bt = blockIdx.x;
  const int D = NF_ * C_;
  const float* row = ph + (long)bt * D;
  float* orow = out + (long)bt * D;
  float s1 = 0.f, s2 = 0.f;
  for (int j = threadIdx.x; j < D; j += 256) { float v = row[j]; s1 += v; s2 += v * v; }
  REDUCE2_256(s1, s2);
  float mean = s1 / D, var = s2 / D - mean * mean;
  float rstd = rsqrtf(var + 1e-5f);
  for (int j = threadIdx.x; j < D; j += 256) {
    float v = (row[j] - mean) * rstd * g[j] + bb[j];
    orow[j] = v + orow[j];
  }
}

// ---------------- LSTM gate kernels ----------------
// freq BiLSTM step s: dir 0 uses position i=s, dir 1 uses i=15-s (output stored
// at position i so the reverse direction lands pre-re-reversed).
__global__ void k_gate_freq(const float* __restrict__ xpf, const float* __restrict__ xpr,
                            const float* __restrict__ gates,
                            float* __restrict__ hst, float* __restrict__ cst,
                            float* __restrict__ yf, float* __restrict__ yr, int s) {
  int idx = blockIdx.x * 256 + threadIdx.x;
  if (idx >= 2 * 2048 * 256) return;
  int dir = idx >> 19;                 // 2048*256 = 2^19
  int rem = idx & 524287;
  int n = rem >> 8, j = rem & 255;
  int i = dir ? (15 - s) : s;
  const float* xp = (dir ? xpr : xpf) + ((long)(n * QD_ + i)) * 1024;
  const float* g = gates + (long)dir * 2048 * 1024 + (long)n * 1024;
  float gi = xp[j] + g[j];
  float gf = xp[256 + j] + g[256 + j];
  float gg = xp[512 + j] + g[512 + j];
  float go = xp[768 + j] + g[768 + j];
  long sidx = (long)dir * 2048 * 256 + (long)n * 256 + j;
  float c = sigf(gf) * cst[sidx] + sigf(gi) * tanhf(gg);
  float h = sigf(go) * tanhf(c);
  cst[sidx] = c; hst[sidx] = h;
  float* y = dir ? yr : yf;
  y[((long)(n * QD_ + i)) * 256 + j] = h;
}
// time LSTM step t; gates already hold x-part + bias + recurrent
__global__ void k_gate_time(const float* __restrict__ gates,
                            float* __restrict__ hst, float* __restrict__ cst,
                            float* __restrict__ ys, int t) {
  int idx = blockIdx.x * 256 + threadIdx.x;
  if (idx >= BNF_ * 256) return;
  int n = idx >> 8, j = idx & 255;
  const float* g = gates + (long)n * 1024;
  float c = sigf(g[256 + j]) * cst[idx] + sigf(g[j]) * tanhf(g[512 + j]);
  float h = sigf(g[768 + j]) * tanhf(c);
  cst[idx] = c; hst[idx] = h;
  ys[((long)n * T_ + t) * (long)H_ + j] = h;
}

// ---------------- scatter / residual / reshape kernels ----------------
// intra = scatter(deconv zz) + deconv_b + x   (pad row f==64 gets zz=0)
__global__ void k_intra(const float* __restrict__ zz, const float* __restrict__ x,
                        const float* __restrict__ db, float* __restrict__ intra) {
  long i = (long)blockIdx.x * 256 + threadIdx.x;
  if (i >= TOT_) return;
  int c = (int)(i % C_);
  long nf = i / C_;
  int f = (int)(nf % NF_);
  long n = nf / NF_;
  float z = 0.f;
  if (f < 64) z = zz[(n * QD_ + (f >> 2)) * 512 + (long)(f & 3) * C_ + c];
  intra[i] = z + db[c] + x[i];
}
// inter[b,t,f,c] = yslin[(b*NF+f)*T + t][c] + intra[b,t,f,c]
__global__ void k_inter(const float* __restrict__ yslin, const float* __restrict__ intra,
                        float* __restrict__ inter) {
  long i = (long)blockIdx.x * 256 + threadIdx.x;
  if (i >= TOT_) return;
  int c = (int)(i % C_);
  long nf = i / C_;
  int f = (int)(nf % NF_);
  long bt = nf / NF_;
  int t = (int)(bt & 511);
  int b = (int)(bt >> 9);
  inter[i] = yslin[(((long)(b * NF_ + f)) * T_ + t) * C_ + c] + intra[i];
}
// avr[b,t,f, nh*VD+v] = av[(b*NH+nh), t, f*VD+v]
__global__ void k_avr(const float* __restrict__ av, float* __restrict__ avr) {
  long i = (long)blockIdx.x * 256 + threadIdx.x;
  if (i >= TOT_) return;
  int c = (int)(i % C_);
  long nf = i / C_;
  int f = (int)(nf % NF_);
  long bt = nf / NF_;
  int t = (int)(bt & 511);
  int b = (int)(bt >> 9);
  int nh = c >> 5, v = c & 31;
  avr[i] = av[(((long)(b * NH_ + nh)) * T_ + t) * (long)(NF_ * VD_) + f * VD_ + v];
}
// K_buf/V_buf -> head of Kf/Vf  (rows < 99)
__global__ void k_copy_kvbuf(const float* __restrict__ src, float* __restrict__ dst,
                             int rows, int D, int Tdst) {
  long i = (long)blockIdx.x * 256 + threadIdx.x;
  long per = (long)rows * D;
  long tot = 16 * per;
  if (i >= tot) return;
  long bh = i / per, rem = i - bh * per;
  dst[bh * (long)Tdst * D + rem] = src[i];
}

// banded softmax over scores rows [bh, t, 0..610]; band s in [t, t+100)
__global__ void k_softmax(float* __restrict__ sc) {
  __shared__ float red[256];
  int bh = blockIdx.x >> 9, t = blockIdx.x & 511;
  float* row = sc + ((long)bh * T_ + t) * (long)S_;
  int tid = threadIdx.x;
  const float scale = rsqrtf((float)(NF_ * E_));    // 1/sqrt(520)
  int lo = t, hi = t + L_;
  float mx = -3.0e38f;
  for (int j = lo + tid; j < hi; j += 256) mx = fmaxf(mx, row[j] * scale);
  red[tid] = mx; __syncthreads();
  for (int o = 128; o > 0; o >>= 1) { if (tid < o) red[tid] = fmaxf(red[tid], red[tid + o]); __syncthreads(); }
  mx = red[0]; __syncthreads();
  float sum = 0.f;
  for (int j = lo + tid; j < hi; j += 256) sum += __expf(row[j] * scale - mx);
  red[tid] = sum; __syncthreads();
  for (int o = 128; o > 0; o >>= 1) { if (tid < o) red[tid] += red[tid + o]; __syncthreads(); }
  float inv = 1.f / red[0];
  for (int j = tid; j < S_; j += 256) {
    float v = 0.f;
    if (j >= lo && j < hi) v = __expf(row[j] * scale - mx) * inv;
    row[j] = v;
  }
}

// ===========================================================================
extern "C" void kernel_launch(void* const* d_in, const int* in_sizes, int n_in,
                              void* d_out, int out_size, void* d_ws, size_t ws_size,
                              hipStream_t stream)
{
  (void)in_sizes; (void)n_in; (void)out_size; (void)ws_size;
  const float* x       = (const float*)d_in[0];
  const float* conv_w  = (const float*)d_in[1];
  const float* conv_b  = (const float*)d_in[2];
  const float* conv_a  = (const float*)d_in[3];
  const float* ln0_g   = (const float*)d_in[4];
  const float* ln0_b   = (const float*)d_in[5];
  const float* iwih_f  = (const float*)d_in[6];
  const float* iwhh_f  = (const float*)d_in[7];
  const float* ibih_f  = (const float*)d_in[8];
  const float* ibhh_f  = (const float*)d_in[9];
  const float* iwih_r  = (const float*)d_in[10];
  const float* iwhh_r  = (const float*)d_in[11];
  const float* ibih_r  = (const float*)d_in[12];
  const float* ibhh_r  = (const float*)d_in[13];
  const float* deconv_w= (const float*)d_in[14];
  const float* deconv_b= (const float*)d_in[15];
  const float* ln1_g   = (const float*)d_in[16];
  const float* ln1_b   = (const float*)d_in[17];
  const float* wih     = (const float*)d_in[18];
  const float* whh     = (const float*)d_in[19];
  const float* bih     = (const float*)d_in[20];
  const float* bhh     = (const float*)d_in[21];
  const float* lin_w   = (const float*)d_in[22];
  const float* lin_b   = (const float*)d_in[23];
  const float* q_w     = (const float*)d_in[24];
  const float* q_b     = (const float*)d_in[25];
  const float* q_a     = (const float*)d_in[26];
  const float* q_lg    = (const float*)d_in[27];
  const float* q_lb    = (const float*)d_in[28];
  const float* k_w     = (const float*)d_in[29];
  const float* k_b     = (const float*)d_in[30];
  const float* k_a     = (const float*)d_in[31];
  const float* k_lg    = (const float*)d_in[32];
  const float* k_lb    = (const float*)d_in[33];
  const float* v_w     = (const float*)d_in[34];
  const float* v_b     = (const float*)d_in[35];
  const float* v_a     = (const float*)d_in[36];
  const float* v_lg    = (const float*)d_in[37];
  const float* v_lb    = (const float*)d_in[38];
  const float* p_w     = (const float*)d_in[39];
  const float* p_b     = (const float*)d_in[40];
  const float* p_a     = (const float*)d_in[41];
  const float* p_lg    = (const float*)d_in[42];
  const float* p_lb    = (const float*)d_in[43];
  const float* K_buf   = (const float*)d_in[44];
  const float* V_buf   = (const float*)d_in[45];
  const float* h0      = (const float*)d_in[46];
  const float* c0      = (const float*)d_in[47];

  float* inter = (float*)d_out;              // d_out doubles as 'inter' buffer

  float* Wk = (float*)d_ws;
  size_t o = 0;
  auto alloc = [&](size_t n) -> float* { float* p = Wk + o; o += n; return p; };
  float* intra = alloc((size_t)TOT_);
  float* buf1  = alloc((size_t)TOT_);        // A_conv -> vtmp -> avr
  float* buf2  = alloc((size_t)TOT_);        // zztmp -> q/k tmp -> ph
  float* hbuf  = alloc((size_t)32768 * 128);
  float* xpf   = alloc((size_t)32768 * 1024);
  float* xpr   = alloc((size_t)32768 * 1024);
  float* ys    = xpf;                        // [260,512,256] reuses xpf(+xpr head)
  float* yf    = alloc((size_t)32768 * 256);
  float* yr    = alloc((size_t)32768 * 256);
  float* fh    = alloc((size_t)2 * 2048 * 256);  // freq h state (contiguous with fc)
  float* fc    = alloc((size_t)2 * 2048 * 256);
  float* th    = alloc((size_t)BNF_ * 256);
  float* tc    = alloc((size_t)BNF_ * 256);
  float* gates = alloc((size_t)2 * 2048 * 1024);
  float* yln   = alloc((size_t)BNF_ * T_ * C_);
  float* yslin = yln;                        // reused after the time-LSTM loop
  float* Qh    = alloc((size_t)16 * 512 * 520);
  float* Kf    = alloc((size_t)16 * 611 * 520);
  float* Vf    = alloc((size_t)16 * 611 * 2080);
  float* sc    = alloc((size_t)16 * 512 * 611);
  float* av    = alloc((size_t)16 * 512 * 2080);
  float* bsum  = alloc((size_t)3072);

  // ---- bf16 weight pool (pre-packed B operands, 16B-aligned sub-regions) ----
  bf16_t* hpool = (bf16_t*)alloc((size_t)800000);
  size_t ho = 0;
  auto halloc = [&](size_t n) -> bf16_t* {
    bf16_t* p = hpool + ho; ho += (n + 7) & ~(size_t)7; return p;
  };
  bf16_t* hWc   = halloc((size_t)128 * 512);
  bf16_t* hWd   = halloc((size_t)2 * 512 * 256);
  bf16_t* hwhhp = halloc((size_t)2 * 1024 * 256);
  bf16_t* hiwf  = halloc((size_t)1024 * 128);
  bf16_t* hiwr  = halloc((size_t)1024 * 128);
  bf16_t* hwih  = halloc((size_t)1024 * 128);
  bf16_t* hwhh  = halloc((size_t)1024 * 256);
  bf16_t* hlin  = halloc((size_t)128 * 256);
  bf16_t* hqw   = halloc((size_t)32 * 128);
  bf16_t* hkw   = halloc((size_t)32 * 128);
  bf16_t* hvw   = halloc((size_t)128 * 128);
  bf16_t* hpw   = halloc((size_t)128 * 128);

  auto ew = [](long n) { return dim3((unsigned)((n + 255) / 256)); };

  // ---------------- packing / constants ----------------
  k_pack_aconv<<<ew((long)32768 * 512), 256, 0, stream>>>(x, buf1);
  k_pack_wc<<<ew(128 * 512), 256, 0, stream>>>(conv_w, hWc);
  k_pack_wd<<<ew(2 * 512 * 256), 256, 0, stream>>>(deconv_w, hWd);
  k_pack_whh<<<ew(2 * 1024 * 256), 256, 0, stream>>>(iwhh_f, iwhh_r, hwhhp);
  k_cvt_bf16<<<ew(1024 * 128), 256, 0, stream>>>(hiwf, iwih_f, 1024 * 128);
  k_cvt_bf16<<<ew(1024 * 128), 256, 0, stream>>>(hiwr, iwih_r, 1024 * 128);
  k_cvt_bf16<<<ew(1024 * 128), 256, 0, stream>>>(hwih, wih, 1024 * 128);
  k_cvt_bf16<<<ew(1024 * 256), 256, 0, stream>>>(hwhh, whh, 1024 * 256);
  k_cvt_bf16<<<ew(128 * 256), 256, 0, stream>>>(hlin, lin_w, 128 * 256);
  k_cvt_bf16<<<ew(32 * 128), 256, 0, stream>>>(hqw, q_w, 32 * 128);
  k_cvt_bf16<<<ew(32 * 128), 256, 0, stream>>>(hkw, k_w, 32 * 128);
  k_cvt_bf16<<<ew(128 * 128), 256, 0, stream>>>(hvw, v_w, 128 * 128);
  k_cvt_bf16<<<ew(128 * 128), 256, 0, stream>>>(hpw, p_w, 128 * 128);
  k_bias_sum<<<ew(3072), 256, 0, stream>>>(ibih_f, ibhh_f, ibih_r, ibhh_r, bih, bhh, bsum);

  // ---------------- intra path ----------------
  // conv (as GEMM) + bias + PReLU, then LN over C
  gemm(buf1, 512, 0, hWc, 512, 0, hbuf, 128, 0, 32768, 128, 512, 1, conv_b, conv_a, 4, stream);
  k_ln_rows<<<32768, 256, 0, stream>>>(hbuf, ln0_g, ln0_b, 128);
  // input projections for both LSTM directions (bias = bih+bhh folded in)
  gemm(hbuf, 128, 0, hiwf, 128, 0, xpf, 1024, 0, 32768, 1024, 128, 1, bsum, nullptr, 3, stream);
  gemm(hbuf, 128, 0, hiwr, 128, 0, xpr, 1024, 0, 32768, 1024, 128, 1, bsum + 1024, nullptr, 3, stream);
  k_fill0<<<ew(2L * 2 * 2048 * 256), 256, 0, stream>>>(fh, 2L * 2 * 2048 * 256);  // fh+fc
  for (int s = 0; s < QD_; ++s) {
    // recurrent projection, both directions batched (grid.z = 2)
    gemm(fh, 256, (long)2048 * 256, hwhhp, 256, (long)1024 * 256,
         gates, 1024, (long)2048 * 1024, 2048, 1024, 256, 2, nullptr, nullptr, 2, stream);
    k_gate_freq<<<4096, 256, 0, stream>>>(xpf, xpr, gates, fh, fc, yf, yr, s);
  }
  // deconv = two GEMMs (yf part, then accumulate yr part)
  gemm(yf, 256, 0, hWd, 256, 0, buf2, 512, 0, 32768, 512, 256, 1, nullptr, nullptr, 2, stream);
  gemm(yr, 256, 0, hWd + (long)512 * 256, 256, 0, buf2, 512, 0, 32768, 512, 256, 1, nullptr, nullptr, 5, stream);
  k_intra<<<ew(TOT_), 256, 0, stream>>>(buf2, x, deconv_b, intra);

  // ---------------- inter path (causal time LSTM) ----------------
  k_ln1t<<<133120, 256, 0, stream>>>(intra, yln, ln1_g, ln1_b);
  k_copy<<<ew((long)BNF_ * 256), 256, 0, stream>>>(th, h0, (long)BNF_ * 256);
  k_copy<<<ew((long)BNF_ * 256), 256, 0, stream>>>(tc, c0, (long)BNF_ * 256);
  for (int t = 0; t < T_; ++t) {
    gemm(yln + (long)t * C_, (long)T_ * C_, 0, hwih, 128, 0, gates, 1024, 0,
         BNF_, 1024, 128, 1, bsum + 2048, nullptr, 3, stream);
    gemm(th, 256, 0, hwhh, 256, 0, gates, 1024, 0, BNF_, 1024, 256, 1, nullptr, nullptr, 5, stream);
    k_gate_time<<<BNF_, 256, 0, stream>>>(gates, th, tc, ys, t);
  }
  gemm(ys, 256, 0, hlin, 256, 0, yslin, 128, 0, BNF_ * T_, 128, 256, 1, lin_b, nullptr, 3, stream);
  k_inter<<<ew(TOT_), 256, 0, stream>>>(yslin, intra, inter);

  // ---------------- attention ----------------
  gemm(inter, 128, 0, hqw, 128, 0, buf2, 32, 0, 133120, 32, 128, 1, q_b, q_a, 4, stream);
  k_ln_head<<<8192, 256, 0, stream>>>(buf2, Qh, q_lg, q_lb, E_, 0, T_);
  gemm(inter, 128, 0, hkw, 128, 0, buf2, 32, 0, 133120, 32, 128, 1, k_b, k_a, 4, stream);
  k_copy_kvbuf<<<ew((long)16 * 99 * 520), 256, 0, stream>>>(K_buf, Kf, L_ - 1, NF_ * E_, S_);
  k_ln_head<<<8192, 256, 0, stream>>>(buf2, Kf, k_lg, k_lb, E_, L_ - 1, S_);
  gemm(inter, 128, 0, hvw, 128, 0, buf1, 128, 0, 133120, 128, 128, 1, v_b, v_a, 4, stream);
  k_copy_kvbuf<<<ew((long)16 * 99 * 2080), 256, 0, stream>>>(V_buf, Vf, L_ - 1, NF_ * VD_, S_);
  k_ln_head<<<8192, 256, 0, stream>>>(buf1, Vf, v_lg, v_lb, VD_, L_ - 1, S_);
  // scores = Qh @ Kf^T (f32 B, batched over 16 heads)
  gemm(Qh, 520, (long)512 * 520, Kf, 520, (long)611 * 520,
       sc, 611, (long)512 * 611, T_, S_, NF_ * E_, 16, nullptr, nullptr, 0, stream);
  k_softmax<<<8192, 256, 0, stream>>>(sc);
  // av = probs @ Vf  (B not transposed)
  gemm(sc, 611, (long)512 * 611, Vf, 2080, (long)611 * 2080,
       av, 2080, (long)512 * 2080, T_, NF_ * VD_, S_, 16, nullptr, nullptr, 1, stream);
  k_avr<<<ew(TOT_), 256, 0, stream>>>(av, buf1);
  gemm(buf1, 128, 0, hpw, 128, 0, buf2, 128, 0, 133120, 128, 128, 1, p_b, p_a, 4, stream);
  // final big LN (8320) + residual into inter (== d_out)
  k_ln_final<<<2048, 256, 0, stream>>>(buf2, inter, p_lg, p_lb);
}
